// InferCellDiscretize_2525440770353
// MI455X (gfx1250) — matrix operand order repaired and checked by reading, not compiled, not run on hardware
//
#include <hip/hip_runtime.h>

typedef __attribute__((ext_vector_type(16))) __bf16 v16bf;
typedef __attribute__((ext_vector_type(8)))  float  v8f;

#define EPS 1e-5f
#define CISTRIDE 72   // padded ci stride (ushorts): 16B-aligned fragments, de-phased banks

// fp32 -> bf16 round-to-nearest-even
__device__ __forceinline__ unsigned short f2bf(float f) {
  unsigned int u = __float_as_uint(f);
  u += 0x7FFFu + ((u >> 16) & 1u);
  return (unsigned short)(u >> 16);
}

__device__ __forceinline__ int argmax8(const float* __restrict__ a) {
  float bv = a[0]; int bi = 0;
#pragma unroll
  for (int i = 1; i < 8; ++i) { float v = a[i]; if (v > bv) { bv = v; bi = i; } }
  return bi;
}

union FragU { uint4 q[2]; v16bf v; };

__device__ __forceinline__ v8f wmma_bf16(v16bf a, v16bf b, v8f c) {
  return __builtin_amdgcn_wmma_f32_16x16x32_bf16(false, a, false, b, (short)0, c,
                                                 false, false);
}

// Async global->LDS copy of 16B, per-lane (CDNA5 ASYNCcnt path).
__device__ __forceinline__ void async_g2l_b128(unsigned ldsAddr,
                                               unsigned long long gaddr) {
  asm volatile("global_load_async_to_lds_b128 %0, %1, off"
               :: "v"(ldsAddr), "v"(gaddr) : "memory");
}
__device__ __forceinline__ void wait_async0() {
  asm volatile("s_wait_asynccnt 0x0" ::: "memory");
}

// ---------------------------------------------------------------------------
// Weight prep: fold BN scale into weights, convert to bf16, pack into WMMA
// B-fragment order: ushort index = ((kc*4 + t)*32 + lane)*16 + q
//   lane<16: N = lane,    K = kc*32 + q        (q = 0..15)
//   lane>=16: N = lane-16, K = kc*32 + 16 + q
// 3x3 K mapping: K = (kh*3+kw)*64 + ci (18 chunks); 1x1: K = ci (2 chunks).
// Also per-edge BN shift sh[e*64+c] = beta - mean*scale.
// ---------------------------------------------------------------------------
__global__ __launch_bounds__(256) void prep_weights_kernel(
    const float* __restrict__ w3, const float* __restrict__ w1,
    const float* __restrict__ bn,
    unsigned short* __restrict__ w3p, unsigned short* __restrict__ w1p,
    float* __restrict__ sh)
{
  const int idx = blockIdx.x * 256 + threadIdx.x;
  if (idx < 110592) {                       // 3 edges * 18 * 4 * 32 * 16
    int u = idx;
    const int q    = u & 15; u >>= 4;
    const int lane = u & 31; u >>= 5;
    const int t    = u & 3;  u >>= 2;
    const int kc   = u % 18;
    const int e    = u / 18;
    const int j    = (lane < 16) ? q : (16 + q);
    const int K    = kc * 32 + j;
    const int pos  = K >> 6;
    const int ci   = K & 63;
    const int n    = t * 16 + (lane & 15);
    const int bnIdx = (e == 0) ? 0 : (e == 1) ? 1 : 3;
    const float g  = bn[(bnIdx * 4 + 0) * 64 + n];
    const float vv = bn[(bnIdx * 4 + 3) * 64 + n];
    const float scale = g / sqrtf(vv + EPS);
    const float val = w3[((size_t)(e * 64 + n) * 64 + ci) * 9 + pos] * scale;
    w3p[idx] = f2bf(val);
  } else if (idx < 118784) {                // 2 edges * 2 * 4 * 32 * 16
    int u = idx - 110592;
    const int q    = u & 15; u >>= 4;
    const int lane = u & 31; u >>= 5;
    const int t    = u & 3;  u >>= 2;
    const int kc   = u & 1;  u >>= 1;
    const int e    = u;
    const int j    = (lane < 16) ? q : (16 + q);
    const int ci   = kc * 32 + j;
    const int n    = t * 16 + (lane & 15);
    const int bnIdx = (e == 0) ? 2 : 4;
    const float g  = bn[(bnIdx * 4 + 0) * 64 + n];
    const float vv = bn[(bnIdx * 4 + 3) * 64 + n];
    const float scale = g / sqrtf(vv + EPS);
    const float val = w1[(size_t)(e * 64 + n) * 64 + ci] * scale;
    w1p[idx - 110592] = f2bf(val);
  } else if (idx < 119104) {                // 5 * 64 BN shifts
    const int k = idx - 118784;
    const int e = k >> 6, c = k & 63;
    const float g  = bn[(e * 4 + 0) * 64 + c];
    const float b  = bn[(e * 4 + 1) * 64 + c];
    const float m  = bn[(e * 4 + 2) * 64 + c];
    const float vv = bn[(e * 4 + 3) * 64 + c];
    sh[k] = b - m * (g / sqrtf(vv + EPS));
  }
}

// ---------------------------------------------------------------------------
// Activation prep for node0: NCHW fp32 -> NHWC bf16 with ReLU + in-mask fused.
// ---------------------------------------------------------------------------
__global__ __launch_bounds__(256) void prep_act_kernel(
    const float* __restrict__ x, const float* __restrict__ alphas1,
    unsigned short* __restrict__ a0)
{
  const int idx = blockIdx.x * 256 + threadIdx.x;   // 32*64*4096 exact
  const int s = idx & 4095;
  const int c = (idx >> 12) & 63;
  const int n = idx >> 18;
  const int c1 = 8 * (argmax8(alphas1) + 1);
  float v = x[idx];
  v = v > 0.0f ? v : 0.0f;
  if (c >= c1) v = 0.0f;
  a0[((size_t)(n << 12) + s) * 64 + c] = f2bf(v);
}

// ---------------------------------------------------------------------------
// Implicit-GEMM conv: one workgroup = 128 pixels (2 full image rows) x 64 Cout.
// Input tile (4 halo rows x 66 halo cols x 64ci, bf16) is staged into LDS with
// async global->LDS b128 copies; halo columns/rows are zero-filled with
// ds_store so the WMMA loop's A-fragment ds_loads are fully unconditional.
// 8 waves; wave w owns 16 pixels, 4 f32 accumulators (Cout tiles of 16).
// Main 3x3 GEMM (K=576) + optional fused 1x1 (K=64, BN scale pre-folded).
// Epilogue: + combined shift, * out-mask; then either bf16 NHWC activation
// (ReLU * in-mask) or final fp32 NCHW store via LDS transpose (aliasing the
// tile buffer) with the skip connection fused.
// ---------------------------------------------------------------------------
__global__ __launch_bounds__(256) void cell_conv_kernel(
    const unsigned short* __restrict__ actA,
    const unsigned short* __restrict__ wpA,
    const unsigned short* __restrict__ actB,
    const unsigned short* __restrict__ wpB,
    const float* __restrict__ sh1,
    const float* __restrict__ sh2,
    const float* __restrict__ alphas1,
    const float* __restrict__ alphas2,
    unsigned short* __restrict__ actOut,
    const float* __restrict__ xSkip,
    float* __restrict__ outFinal,
    int hasB, int finalMode)
{
  // 4*66*CISTRIDE ushorts = 38016 B; final transpose (128*65 f32 = 33280 B) aliases it
  __shared__ __align__(16) char smemRaw[4 * 66 * CISTRIDE * 2];
  unsigned short* tile = (unsigned short*)smemRaw;
  float* xpose = (float*)smemRaw;

  const int tid  = threadIdx.x;
  const int wave = tid >> 5;
  const int lane = tid & 31;
  const int lm   = lane & 15;
  const int hi   = lane >> 4;
  const int tileBase = blockIdx.x * 128;       // global pixel of tile start
  const int img  = tileBase >> 12;             // image index
  const int r0   = (tileBase & 4095) >> 6;     // first output row in image

  if (tid == 0) __builtin_prefetch(wpA, 0, 1); // pull weight panel toward L2

  // ---- stage input tile: 4 rows x 66 cols, 8 x b128 chunks per (row,col) ----
  const unsigned ldsBase = (unsigned)(unsigned long long)(void*)tile;
  const uint4 z4 = make_uint4(0u, 0u, 0u, 0u);
#pragma unroll
  for (int it = 0; it < 9; ++it) {
    const int idx = it * 256 + tid;            // < 4*66*8 = 2112
    if (idx < 2112) {
      const int trow = idx / 528;
      const int rem  = idx - trow * 528;
      const int col  = rem >> 3;
      const int ck   = rem & 7;
      const unsigned lOff = ((unsigned)(trow * 66 + col) * CISTRIDE + ck * 8) * 2;
      const int r = r0 - 1 + trow;             // input image row
      if (col == 0 || col == 65 || r < 0 || r > 63) {
        *(uint4*)(smemRaw + lOff) = z4;        // zero halo (ds_store_b128)
      } else {
        const size_t gOff =
            ((size_t)((img << 12) + r * 64 + (col - 1)) * 64 + ck * 8) * 2;
        async_g2l_b128(ldsBase + lOff,
                       (unsigned long long)(const char*)actA + gOff);
      }
    }
  }
  wait_async0();        // this wave's async copies landed in LDS
  __syncthreads();      // everyone's copies/zero-fills visible

  const int p    = wave * 16 + lm;             // local pixel (A-matrix row)
  const int lrow = p >> 6;                     // 0 or 1
  const int lcol = p & 63;

  v8f acc[4] = {};

  // ---- main 3x3 conv: 18 K-chunks of 32, A from LDS, B from global/L2 ----
#pragma unroll
  for (int pos = 0; pos < 9; ++pos) {
    const int trow = lrow + pos / 3;           // 0..3
    const int tcol = lcol + pos % 3;           // 0..65
    const unsigned abase = (unsigned)(trow * 66 + tcol) * CISTRIDE + hi * 16;
#pragma unroll
    for (int cih = 0; cih < 2; ++cih) {
      FragU A;
      const uint4* pa = (const uint4*)&tile[abase + cih * 32];
      A.q[0] = pa[0];      // K run {0..7} (lo half) / {16..23} (hi half)
      A.q[1] = pa[1];      // K run {8..15} / {24..31}
      const int kc = pos * 2 + cih;
#pragma unroll
      for (int t = 0; t < 4; ++t) {
        FragU B;
        const uint4* pb =
            (const uint4*)(wpA + ((size_t)((kc * 4 + t) * 32 + lane)) * 16);
        B.q[0] = pb[0]; B.q[1] = pb[1];
        acc[t] = wmma_bf16(A.v, B.v, acc[t]);
      }
    }
  }

  // ---- optional fused 1x1 conv: 2 K-chunks of 32 (exact-pixel, no halo) ----
  if (hasB) {
    const unsigned short* ab0 = actB + (size_t)(tileBase + p) * 64 + hi * 16;
#pragma unroll
    for (int kc = 0; kc < 2; ++kc) {
      FragU A;
      const uint4* pa = (const uint4*)(ab0 + kc * 32);
      A.q[0] = pa[0];
      A.q[1] = pa[1];
#pragma unroll
      for (int t = 0; t < 4; ++t) {
        FragU B;
        const uint4* pb =
            (const uint4*)(wpB + ((size_t)((kc * 4 + t) * 32 + lane)) * 16);
        B.q[0] = pb[0]; B.q[1] = pb[1];
        acc[t] = wmma_bf16(A.v, B.v, acc[t]);
      }
    }
  }

  // ---- epilogue ----
  const int c1 = 8 * (argmax8(alphas1) + 1);
  const int c2 = 8 * (argmax8(alphas2) + 1);

  if (finalMode) __syncthreads();   // tile buffer is dead; safe to alias

#pragma unroll
  for (int t = 0; t < 4; ++t) {
    const int c = t * 16 + lm;
    const float shc = sh1[c] + (sh2 ? sh2[c] : 0.0f);
    const float om = (c < c2) ? 1.0f : 0.0f;
#pragma unroll
    for (int v = 0; v < 8; ++v) {
      const int pl = wave * 16 + v + hi * 8;          // local pixel for VGPR v
      const float val = (acc[t][v] + shc) * om;
      if (finalMode) {
        xpose[pl * 65 + c] = val;
      } else {
        float av = val > 0.0f ? val : 0.0f;           // ReLU for next edge
        if (c >= c1) av = 0.0f;                       // next edge's in-mask
        actOut[(size_t)(tileBase + pl) * 64 + c] = f2bf(av);
      }
    }
  }

  if (finalMode) {
    __syncthreads();
    const int rowPix = tileBase & 4095;
#pragma unroll 4
    for (int it = 0; it < 32; ++it) {
      const int idx = it * 256 + tid;     // 0..8191
      const int c = idx >> 7;
      const int pp = idx & 127;
      const size_t o = ((size_t)(img * 64 + c)) * 4096 + rowPix + pp;
      outFinal[o] = xpose[pp * 65 + c] + xSkip[o];    // skip connection (n0)
    }
  }
}

// ---------------------------------------------------------------------------
extern "C" void kernel_launch(void* const* d_in, const int* in_sizes, int n_in,
                              void* d_out, int out_size, void* d_ws, size_t ws_size,
                              hipStream_t stream) {
  const float* x   = (const float*)d_in[0];
  const float* al1 = (const float*)d_in[1];
  const float* al2 = (const float*)d_in[2];
  const float* w3  = (const float*)d_in[3];
  const float* w1  = (const float*)d_in[4];
  const float* bn  = (const float*)d_in[5];
  float* out = (float*)d_out;

  char* ws = (char*)d_ws;
  unsigned short* a0  = (unsigned short*)ws; ws += 16777216;  // 32*4096*64 bf16
  unsigned short* a1  = (unsigned short*)ws; ws += 16777216;
  unsigned short* a2  = (unsigned short*)ws; ws += 16777216;
  unsigned short* w3p = (unsigned short*)ws; ws += 221184;    // 3*18*4*32*16*2
  unsigned short* w1p = (unsigned short*)ws; ws += 16384;     // 2*2*4*32*16*2
  float* sh = (float*)ws;                                     // 5*64 fp32

  prep_weights_kernel<<<466, 256, 0, stream>>>(w3, w1, bn, w3p, w1p, sh);
  prep_act_kernel<<<32768, 256, 0, stream>>>(x, al1, a0);

  const int W3E = 18 * 4 * 32 * 16;   // ushorts per 3x3 edge
  const int W1E = 2 * 4 * 32 * 16;    // ushorts per 1x1 edge

  // node1 = BN0(conv3(a0)) -> a1 (activated bf16)
  cell_conv_kernel<<<1024, 256, 0, stream>>>(
      a0, w3p + 0 * W3E, nullptr, nullptr, sh + 0, nullptr,
      al1, al2, a1, nullptr, nullptr, 0, 0);
  // node2 = BN1(conv3(a0)) + BN2(conv1(a1)) -> a2
  cell_conv_kernel<<<1024, 256, 0, stream>>>(
      a0, w3p + 1 * W3E, a1, w1p + 0 * W1E, sh + 64, sh + 128,
      al1, al2, a2, nullptr, nullptr, 1, 0);
  // node3 = n0 + BN3(conv3(a1)) + BN4(conv1(a2)) -> out (NCHW fp32)
  cell_conv_kernel<<<1024, 256, 0, stream>>>(
      a1, w3p + 2 * W3E, a2, w1p + 1 * W1E, sh + 192, sh + 256,
      al1, al2, nullptr, x, out, 1, 1);
}
